// FlashAttention_67834713473676
// MI455X (gfx1250) — compile-verified, matrix operand order checked by reading, not run on hardware
//
#include <hip/hip_runtime.h>

typedef __attribute__((ext_vector_type(16))) _Float16 v16h;
typedef __attribute__((ext_vector_type(8)))  _Float16 v8h;
typedef __attribute__((ext_vector_type(4)))  _Float16 v4h;
typedef __attribute__((ext_vector_type(8)))  float    v8f;
typedef __attribute__((ext_vector_type(4)))  int      v4i;

#define SEQ     2048
#define DHEAD   64
#define BM      64      // query rows per block (16 per wave)
#define BN      64      // keys per iteration
#define KSTRIDE 72      // halves per K-LDS row (64 + pad, 144B, 16B-aligned)
#define VSTRIDE 72      // halves per Vt-LDS row (64 + pad, 144B, 16B-aligned)

// 1/sqrt(64) * log2(e): scores produced directly in log2 domain
#define QSCALE  (0.125f * 1.4426950408889634f)

#if __has_builtin(__builtin_amdgcn_global_load_async_to_lds_b128)
#define ASYNC_STAGE 1
#else
#define ASYNC_STAGE 0
#endif

// Assemble a v16h WMMA operand from two contiguous 8-half runs at p and p+16,
// matching the 16-bit A/B VGPR layout: elem e -> K = kbase + (e&7) + (e>=8)*16.
__device__ __forceinline__ v16h ld_op16(const _Float16* p) {
    const v8h lo = *(const v8h*)(p);
    const v8h hi = *(const v8h*)(p + 16);
    v16h r;
#pragma unroll
    for (int i = 0; i < 8; ++i) { r[i] = lo[i]; r[i + 8] = hi[i]; }
    return r;
}

__device__ __forceinline__ v8f wmma_f16(v16h a, v16h b, v8f c) {
    return __builtin_amdgcn_wmma_f32_16x16x32_f16(
        /*neg_a=*/false, a, /*neg_b=*/false, b,
        /*c_mod=*/(short)0, c, /*reuse_a=*/false, /*reuse_b=*/false);
}

// Stage one 64(key) x 64(d) K block and one 64(d) x 64(key) Vt block into LDS.
__device__ __forceinline__ void stage_block(const _Float16* kg, const _Float16* vtg,
                                            _Float16* ks, _Float16* vts, int tid) {
#pragma unroll
    for (int i = 0; i < 4; ++i) {
        const int c   = tid + 128 * i;  // 0..511 16B chunks
        const int row = c >> 3;
        const int off = (c & 7) * 8;
#if ASYNC_STAGE
        __builtin_amdgcn_global_load_async_to_lds_b128(
            (v4i*)(kg + row * DHEAD + off),
            (v4i*)&ks[row * KSTRIDE + off], 0, 0);
        __builtin_amdgcn_global_load_async_to_lds_b128(
            (v4i*)(vtg + (size_t)row * SEQ + off),
            (v4i*)&vts[row * VSTRIDE + off], 0, 0);
#else
        *(v8h*)&ks[row * KSTRIDE + off]  = *(const v8h*)(kg + row * DHEAD + off);
        *(v8h*)&vts[row * VSTRIDE + off] = *(const v8h*)(vtg + (size_t)row * SEQ + off);
#endif
    }
}

// One 64-key online-softmax step against the LDS block (ksb, vtsb).
__device__ __forceinline__ void fa_step(const _Float16* ksb, const _Float16* vtsb,
                                        const v16h (&qb)[2], const v16h& ones,
                                        int l16, int kbase,
                                        float& m_i, float& l_i, v8f (&acc)[4]) {
    // ---- S^T = K Q^T : four 16(key)x16(qrow) tiles, k=64 in two chunks ----
    v8f s[4];
#pragma unroll
    for (int n = 0; n < 4; ++n)
#pragma unroll
        for (int v = 0; v < 8; ++v) s[n][v] = 0.0f;
#pragma unroll
    for (int c = 0; c < 2; ++c) {
        const int dco = c * 32 + kbase;
#pragma unroll
        for (int n = 0; n < 4; ++n) {
            const v16h ka = ld_op16(&ksb[(n * 16 + l16) * KSTRIDE + dco]);
            s[n] = wmma_f16(ka, qb[c], s[n]);
        }
    }

    // ---- online softmax in log2 domain; balanced max tree ----
    float tm[4];
#pragma unroll
    for (int n = 0; n < 4; ++n) {
        const float a0 = fmaxf(s[n][0], s[n][1]);
        const float a1 = fmaxf(s[n][2], s[n][3]);
        const float a2 = fmaxf(s[n][4], s[n][5]);
        const float a3 = fmaxf(s[n][6], s[n][7]);
        tm[n] = fmaxf(fmaxf(a0, a1), fmaxf(a2, a3));
    }
    float mx = fmaxf(fmaxf(tm[0], tm[1]), fmaxf(tm[2], tm[3]));
    mx = fmaxf(mx, __shfl_xor(mx, 16, 32));

    const float mnew = fmaxf(m_i, mx);
    const float corr = __builtin_amdgcn_exp2f(m_i - mnew);

    v16h pb[2];  // P^T packed directly into B-operand layout
#pragma unroll
    for (int n = 0; n < 4; ++n)
#pragma unroll
        for (int v = 0; v < 8; ++v)
            pb[n >> 1][(n & 1) * 8 + v] =
                (_Float16)__builtin_amdgcn_exp2f(s[n][v] - mnew);

    // row sum via WMMA with ones: every lane gets sum over all 64 keys
    v8f rsum;
#pragma unroll
    for (int v = 0; v < 8; ++v) rsum[v] = 0.0f;
    rsum = wmma_f16(ones, pb[0], rsum);
    rsum = wmma_f16(ones, pb[1], rsum);

    l_i = l_i * corr + rsum[0];
    m_i = mnew;
#pragma unroll
    for (int t = 0; t < 4; ++t)
#pragma unroll
        for (int v = 0; v < 8; ++v) acc[t][v] *= corr;

    // ---- O^T += V^T P^T : 4 d-tiles x 2 key-chunks ----
#pragma unroll
    for (int ck = 0; ck < 2; ++ck)
#pragma unroll
        for (int t = 0; t < 4; ++t) {
            const v16h va = ld_op16(&vtsb[(t * 16 + l16) * VSTRIDE + ck * 32 + kbase]);
            acc[t] = wmma_f16(va, pb[ck], acc[t]);
        }
}

// ---------------------------------------------------------------------------
// Prep: one-time f32 -> f16 conversion into workspace.
//   Qh[bh][row][d] = Q * QSCALE   (row-major)
//   Kh[bh][key][d] = K            (row-major)
//   Vth[bh][d][key] = V           (transposed, d-major)
// ---------------------------------------------------------------------------
__global__ __launch_bounds__(256)
void fa_prep(const float* __restrict__ Q, const float* __restrict__ K,
             const float* __restrict__ V, _Float16* __restrict__ Qh,
             _Float16* __restrict__ Kh, _Float16* __restrict__ Vth) {
    const size_t i = ((size_t)blockIdx.x * blockDim.x + threadIdx.x) * 4;

    const float4 q4 = *(const float4*)(Q + i);
    const float4 k4 = *(const float4*)(K + i);
    const float4 v4 = *(const float4*)(V + i);

    v4h qh, kh;
    qh[0] = (_Float16)(q4.x * QSCALE); qh[1] = (_Float16)(q4.y * QSCALE);
    qh[2] = (_Float16)(q4.z * QSCALE); qh[3] = (_Float16)(q4.w * QSCALE);
    kh[0] = (_Float16)k4.x; kh[1] = (_Float16)k4.y;
    kh[2] = (_Float16)k4.z; kh[3] = (_Float16)k4.w;
    *(v4h*)(Qh + i) = qh;
    *(v4h*)(Kh + i) = kh;

    // scatter V transposed
    const size_t bh  = i / (SEQ * DHEAD);
    const int    rem = (int)(i - bh * (SEQ * DHEAD));
    const int    key = rem >> 6;
    const int    d0  = rem & 63;
    _Float16* vtb = Vth + bh * (SEQ * DHEAD);
    vtb[(size_t)(d0 + 0) * SEQ + key] = (_Float16)v4.x;
    vtb[(size_t)(d0 + 1) * SEQ + key] = (_Float16)v4.y;
    vtb[(size_t)(d0 + 2) * SEQ + key] = (_Float16)v4.z;
    vtb[(size_t)(d0 + 3) * SEQ + key] = (_Float16)v4.w;
}

// ---------------------------------------------------------------------------
// Flash attention forward: S^T / O^T formulation, double-buffered async LDS,
// hand-unrolled x2 so both LDS buffers are compile-time bases.
// ---------------------------------------------------------------------------
__global__ __launch_bounds__(128)
void fa_fwd_wmma(const _Float16* __restrict__ Qh, const _Float16* __restrict__ Kh,
                 const _Float16* __restrict__ Vth, float* __restrict__ O) {
    __shared__ __align__(16) _Float16 Ks0[BN * KSTRIDE];
    __shared__ __align__(16) _Float16 Ks1[BN * KSTRIDE];
    __shared__ __align__(16) _Float16 Vts0[DHEAD * VSTRIDE];
    __shared__ __align__(16) _Float16 Vts1[DHEAD * VSTRIDE];

    const int tid   = threadIdx.x;
    const int lane  = tid & 31;
    const int wid   = tid >> 5;
    const int half  = lane >> 4;
    const int l16   = lane & 15;
    const int kbase = half * 8;

    const int    bh     = blockIdx.y;
    const int    qblk   = blockIdx.x;
    const size_t bh_off = (size_t)bh * SEQ * DHEAD;

    const _Float16* kg0  = Kh  + bh_off;
    const _Float16* vtg0 = Vth + bh_off;

    // ---- Q tile: B-operand (Q^T) registers, straight f16 loads ----
    const int       qbase = qblk * BM + wid * 16;
    const _Float16* qrow  = Qh + bh_off + (size_t)(qbase + l16) * DHEAD;
    v16h qb[2];
#pragma unroll
    for (int c = 0; c < 2; ++c) qb[c] = ld_op16(qrow + c * 32 + kbase);

    v16h ones;
#pragma unroll
    for (int e = 0; e < 16; ++e) ones[e] = (_Float16)1.0f;

    float m_i = -1e30f, l_i = 0.0f;  // per-lane: this lane owns qrow (qbase+l16)
    v8f acc[4];                      // O^T: rows d = t*16 + kbase + v, col = qrow
#pragma unroll
    for (int t = 0; t < 4; ++t)
#pragma unroll
        for (int v = 0; v < 8; ++v) acc[t][v] = 0.0f;

#if ASYNC_STAGE
#define STAGE_WAIT() asm volatile("s_wait_asynccnt 0" ::: "memory")
#else
#define STAGE_WAIT()
#endif

    // prologue: prefetch first block into buffer 0
    stage_block(kg0, vtg0, Ks0, Vts0, tid);

    for (int kb = 0; kb < SEQ; kb += 2 * BN) {
        // ---- half A: compute buffer 0, prefetch buffer 1 ----
        STAGE_WAIT();
        __syncthreads();  // buffer 0 ready everywhere; buffer 1 free
        // kb + BN < SEQ always holds (SEQ/BN even)
        stage_block(kg0 + (size_t)(kb + BN) * DHEAD, vtg0 + (kb + BN),
                    Ks1, Vts1, tid);
        fa_step(Ks0, Vts0, qb, ones, l16, kbase, m_i, l_i, acc);

        // ---- half B: compute buffer 1, prefetch buffer 0 ----
        STAGE_WAIT();
        __syncthreads();  // buffer 1 ready everywhere; buffer 0 free
        if (kb + 2 * BN < SEQ)
            stage_block(kg0 + (size_t)(kb + 2 * BN) * DHEAD, vtg0 + (kb + 2 * BN),
                        Ks0, Vts0, tid);
        fa_step(Ks1, Vts1, qb, ones, l16, kbase, m_i, l_i, acc);
    }

    // ---- epilogue: O[qrow][d] = acc^T / l, b128 runs along d ----
    const float inv  = 1.0f / l_i;
    float*      orow = O + bh_off + (size_t)(qbase + l16) * DHEAD;
#pragma unroll
    for (int t = 0; t < 4; ++t) {
        float4 lo, hi;
        lo.x = acc[t][0] * inv; lo.y = acc[t][1] * inv;
        lo.z = acc[t][2] * inv; lo.w = acc[t][3] * inv;
        hi.x = acc[t][4] * inv; hi.y = acc[t][5] * inv;
        hi.z = acc[t][6] * inv; hi.w = acc[t][7] * inv;
        *(float4*)&orow[t * 16 + kbase]     = lo;
        *(float4*)&orow[t * 16 + kbase + 4] = hi;
    }
}

extern "C" void kernel_launch(void* const* d_in, const int* in_sizes, int n_in,
                              void* d_out, int out_size, void* d_ws, size_t ws_size,
                              hipStream_t stream) {
    const float* Q = (const float*)d_in[0];
    const float* K = (const float*)d_in[1];
    const float* V = (const float*)d_in[2];
    float*       O = (float*)d_out;

    const size_t n  = (size_t)in_sizes[0];        // B*H*S*D = 4,194,304
    const int    bh = (int)(n / (SEQ * DHEAD));   // 32

    _Float16* Qh  = (_Float16*)d_ws;
    _Float16* Kh  = Qh + n;
    _Float16* Vth = Kh + n;

    fa_prep<<<dim3((unsigned)(n / 4 / 256)), dim3(256), 0, stream>>>(Q, K, V, Qh, Kh, Vth);

    dim3 grid(SEQ / BM, bh);
    fa_fwd_wmma<<<grid, dim3(128), 0, stream>>>(Qh, Kh, Vth, O);
}
